// FixedGCNN_49323404427580
// MI455X (gfx1250) — matrix operand order repaired and checked by reading, not compiled, not run on hardware
//
#include <hip/hip_runtime.h>
#include <hip/hip_bf16.h>

// ---------------------------------------------------------------------------
// DGCNN segmentation network on gfx1250 (MI455X).
// All GEMMs run through v_wmma_f32_16x16x32_f16. Each wave32 computes a
// 32x16 output block (two 16x16 tiles sharing one B fragment) so every
// weight-fragment load feeds two WMMAs. BatchNorm (training-mode batch
// stats): GEMM epilogue -> atomic per-channel sum/sumsq -> finalize ->
// affine+LeakyReLU fused into the consumer's loads. The fused affine runs
// entirely in packed f16 (lrelu(x) == max(x, 0.2x)) so the A-side epilogue
// is fma/mul/max 16-bit VALU that can co-issue with the XDL WMMA pipe.
// ---------------------------------------------------------------------------

typedef __attribute__((ext_vector_type(16))) _Float16 v16h;
typedef __attribute__((ext_vector_type(8)))  _Float16 v8h;
typedef __attribute__((ext_vector_type(8)))  float    v8f;

#define NPTS   65536
#define KNBR   16
#define NKROWS ((size_t)NPTS * KNBR)
#define NSEG   4
#define SEGSZ  (NPTS / NSEG)

__device__ __forceinline__ float lrelu(float v) {
    return fmaxf(v, 0.2f * v);          // LeakyReLU(0.2) == max(x, 0.2x)
}

__device__ __forceinline__ _Float16 lrelu_h(_Float16 v) {
    _Float16 z = v * (_Float16)0.2f;
#if __has_builtin(__builtin_fmaxf16)
    return __builtin_fmaxf16(v, z);
#else
    return v > z ? v : z;
#endif
}

// ---------------------------------------------------------------------------
// Weight convert: fp32 [Or, Ir] -> f16 [Op, Ip], zero padded.
// ---------------------------------------------------------------------------
__global__ void wcvt_kernel(const float* __restrict__ W, _Float16* __restrict__ Wh,
                            int Or, int Ir, int Op, int Ip) {
    int idx = blockIdx.x * blockDim.x + threadIdx.x;
    int total = Op * Ip;
    if (idx >= total) return;
    int o = idx / Ip, i = idx - o * Ip;
    float v = (o < Or && i < Ir) ? W[(size_t)o * Ir + i] : 0.0f;
    Wh[idx] = (_Float16)v;
}

// ---------------------------------------------------------------------------
// Edge feature build, first layer: feats [N,6] fp32 -> E [N*K, 32] f16
// cols 0..5 = center, 6..11 = neighbor-center, 12..31 = 0 (pad to K-dim 32).
// ---------------------------------------------------------------------------
__global__ void gather_first_kernel(const float* __restrict__ feats,
                                    const long long* __restrict__ nidx,
                                    _Float16* __restrict__ E) {
    int row = blockIdx.x * blockDim.x + threadIdx.x;   // n*K + j
    if (row >= (int)NKROWS) return;
    int n = row / KNBR;
    long long nb = nidx[row];
    _Float16* er = E + (size_t)row * 32;
    float c[6];
#pragma unroll
    for (int i = 0; i < 6; ++i) c[i] = feats[(size_t)n * 6 + i];
#pragma unroll
    for (int i = 0; i < 6; ++i) {
        er[i]     = (_Float16)c[i];
        er[6 + i] = (_Float16)(feats[(size_t)nb * 6 + i] - c[i]);
    }
#pragma unroll
    for (int i = 12; i < 32; ++i) er[i] = (_Float16)0.0f;
}

// ---------------------------------------------------------------------------
// Edge feature build, C=64 layers: x [N,64] f16 -> E [N*K, 128] f16
// ---------------------------------------------------------------------------
__global__ void gather_edge64_kernel(const _Float16* __restrict__ x,
                                     const long long* __restrict__ nidx,
                                     _Float16* __restrict__ E) {
    int t = blockIdx.x * blockDim.x + threadIdx.x;     // (row, 8-col chunk)
    int total = (int)(NKROWS * 8);
    if (t >= total) return;
    int row = t >> 3;
    int c8  = (t & 7) * 8;
    int n = row / KNBR;
    long long nb = nidx[row];
    const _Float16* xc = x + (size_t)n  * 64 + c8;
    const _Float16* xn = x + (size_t)nb * 64 + c8;
    _Float16* er = E + (size_t)row * 128;
#pragma unroll
    for (int i = 0; i < 8; ++i) {
        _Float16 cv = xc[i];
        er[c8 + i]      = cv;
        er[64 + c8 + i] = (_Float16)((float)xn[i] - (float)cv);
    }
}

// ---------------------------------------------------------------------------
// A-fragment loader (16x32 f16, CDNA5 ISA 7.12.2 layout):
//   lane<16 : row M=lane,    K in {hi*8..hi*8+7, 16+hi*8..+7} with hi=0
//   lane>=16: row M=lane-16, same chunks with hi=1
// Optionally applies per-K-channel BN affine + LeakyReLU (previous layer),
// entirely in f16 (scale/shift pre-converted fragments passed in).
// ---------------------------------------------------------------------------
template <bool AFFINE>
__device__ __forceinline__ v16h load_a_frag(const _Float16* __restrict__ Abase,
                                            int kt, int hi,
                                            v8h sc0, v8h sh0, v8h sc1, v8h sh1) {
    v8h a0 = *(const v8h*)(Abase + kt + hi * 8);
    v8h a1 = *(const v8h*)(Abase + kt + 16 + hi * 8);
    v16h af;
    if (AFFINE) {
#pragma unroll
        for (int i = 0; i < 8; ++i) {
            _Float16 y0 = a0[i] * sc0[i] + sh0[i];   // v_pk_fma_f16 class
            _Float16 y1 = a1[i] * sc1[i] + sh1[i];
            af[i]     = lrelu_h(y0);                 // v_pk_max_num_f16 class
            af[8 + i] = lrelu_h(y1);
        }
    } else {
#pragma unroll
        for (int i = 0; i < 8; ++i) { af[i] = a0[i]; af[8 + i] = a1[i]; }
    }
    return af;
}

// ---------------------------------------------------------------------------
// WMMA GEMM:  D[M,O] = act(A) [M,Kc] f16  x  W[O,Kc]^T f16   (f32 accumulate)
//   - one wave32 per 32x16 output block: two 16x16 tiles share one B fragment
//   - templated fused per-input-channel f16 affine + LeakyReLU on A
//   - epilogue: f16 store + atomic per-channel sum/sumsq (for BN), or fp32
//     store clipped to Oreal columns (final layer).
// ---------------------------------------------------------------------------
template <bool AFFINE>
__launch_bounds__(256)
__global__ void wmma_gemm_kernel(const _Float16* __restrict__ A,
                                 const _Float16* __restrict__ W,
                                 _Float16* __restrict__ Ch,
                                 float* __restrict__ outF,
                                 float* __restrict__ sum,
                                 float* __restrict__ sumsq,
                                 const _Float16* __restrict__ a_scale_h,
                                 const _Float16* __restrict__ a_shift_h,
                                 int M, int Kc, int O, int Oreal) {
    const int wave = threadIdx.x >> 5;
    const int lane = threadIdx.x & 31;
    const int tiles_o = O >> 4;
    const long long blk = (long long)blockIdx.x * 8 + wave;      // 32-row block
    const long long total_blks = (long long)(M >> 5) * tiles_o;
    if (blk >= total_blks) return;          // wave-uniform: EXEC stays all-1s
    const int ot = (int)(blk % tiles_o);
    const int mt = (int)(blk / tiles_o);
    const int m_base = mt << 5, o_base = ot << 4;
    const int hi = lane >> 4, lq = lane & 15;

    const _Float16* Abase0 = A + (size_t)(m_base + lq) * Kc;        // tile 0
    const _Float16* Abase1 = A + (size_t)(m_base + 16 + lq) * Kc;   // tile 1
    const _Float16* Wbase  = W + (size_t)(o_base + lq) * Kc + hi * 8;

    v8f acc0 = {}, acc1 = {};
#pragma unroll 2
    for (int kt = 0; kt < Kc; kt += 32) {
        v8h sc0 = {}, sh0 = {}, sc1 = {}, sh1 = {};
        if (AFFINE) {      // scale/shift fragments, same K layout as A chunks
            sc0 = *(const v8h*)(a_scale_h + kt + hi * 8);
            sh0 = *(const v8h*)(a_shift_h + kt + hi * 8);
            sc1 = *(const v8h*)(a_scale_h + kt + 16 + hi * 8);
            sh1 = *(const v8h*)(a_shift_h + kt + 16 + hi * 8);
        }
        v16h af0 = load_a_frag<AFFINE>(Abase0, kt, hi, sc0, sh0, sc1, sh1);
        v16h af1 = load_a_frag<AFFINE>(Abase1, kt, hi, sc0, sh0, sc1, sh1);
        // B 32x16 f16 fragment: lane holds K = {hi*8..hi*8+7, 16+hi*8..+7}, col = lq
        v8h b0 = *(const v8h*)(Wbase + kt);
        v8h b1 = *(const v8h*)(Wbase + kt + 16);
        v16h bf = __builtin_shufflevector(b0, b1, 0, 1, 2, 3, 4, 5, 6, 7, 8, 9, 10, 11, 12, 13, 14, 15);
        if (kt + 32 < Kc) {  // CDNA5 global_prefetch_b8 of next K-slab
            __builtin_prefetch((const void*)(Abase0 + kt + 32), 0, 1);
            __builtin_prefetch((const void*)(Abase1 + kt + 32), 0, 1);
        }
        acc0 = __builtin_amdgcn_wmma_f32_16x16x32_f16(
                   false, af0, false, bf, (short)0, acc0, false, false);
        acc1 = __builtin_amdgcn_wmma_f32_16x16x32_f16(
                   false, af1, false, bf, (short)0, acc1, false, false);
    }

    const int col = o_base + lq;
    if (outF) {                      // final layer: fp32 out, O padded -> Oreal
        if (col < Oreal) {
#pragma unroll
            for (int r = 0; r < 8; ++r) {
                int m = m_base + r + hi * 8;     // D layout: M = r + 8*hi
                outF[(size_t)m * Oreal + col]        = acc0[r];
                outF[(size_t)(m + 16) * Oreal + col] = acc1[r];
            }
        }
    } else {
        float s = 0.0f, s2 = 0.0f;
#pragma unroll
        for (int r = 0; r < 8; ++r) {
            int m = m_base + r + hi * 8;
            float v0 = acc0[r], v1 = acc1[r];
            Ch[(size_t)m * O + col]        = (_Float16)v0;
            Ch[(size_t)(m + 16) * O + col] = (_Float16)v1;
            s += v0 + v1; s2 += v0 * v0 + v1 * v1;
        }
        if (sum) {
            atomicAdd(&sum[col], s);
            atomicAdd(&sumsq[col], s2);
        }
    }
}

// ---------------------------------------------------------------------------
// BN finalize: per-channel scale/shift from batch sum/sumsq (biased var).
// Emits both f32 (for f32 consumers) and f16 (for fused GEMM A-path) tables.
// ---------------------------------------------------------------------------
__global__ void bn_finalize_kernel(const float* __restrict__ sum,
                                   const float* __restrict__ sumsq,
                                   const float* __restrict__ gamma,
                                   const float* __restrict__ beta,
                                   float* __restrict__ scale,
                                   float* __restrict__ shift,
                                   _Float16* __restrict__ scale_h,
                                   _Float16* __restrict__ shift_h,
                                   int O, float invM) {
    int c = blockIdx.x * blockDim.x + threadIdx.x;
    if (c >= O) return;
    float m  = sum[c] * invM;
    float v  = fmaf(sumsq[c], invM, -m * m);
    float r  = rsqrtf(v + 1e-5f);
    float sc = gamma[c] * r;
    float sh = beta[c] - m * sc;
    scale[c] = sc;
    shift[c] = sh;
    scale_h[c] = (_Float16)sc;
    shift_h[c] = (_Float16)sh;
}

// ---------------------------------------------------------------------------
// Apply BN-affine + LeakyReLU then max over K neighbors: [N*K,C] -> [N,C] f16
// ---------------------------------------------------------------------------
__global__ void maxk_apply_kernel(const _Float16* __restrict__ R,
                                  const float* __restrict__ scale,
                                  const float* __restrict__ shift,
                                  _Float16* __restrict__ xout, int C) {
    int t = blockIdx.x * blockDim.x + threadIdx.x;
    if (t >= NPTS * C) return;
    int n = t / C, c = t - n * C;
    float sc = scale[c], sh = shift[c];
    float mx = -3.4e38f;
    const _Float16* base = R + (size_t)n * KNBR * C + c;
#pragma unroll
    for (int j = 0; j < KNBR; ++j) {
        float v = lrelu(fmaf((float)base[(size_t)j * C], sc, sh));
        mx = fmaxf(mx, v);
    }
    xout[t] = (_Float16)mx;
}

// ---------------------------------------------------------------------------
// concat [x1|x2|x3] -> [N,192] f16
// ---------------------------------------------------------------------------
__global__ void concat3_kernel(const _Float16* __restrict__ x1,
                               const _Float16* __restrict__ x2,
                               const _Float16* __restrict__ x3,
                               _Float16* __restrict__ out) {
    int t = blockIdx.x * blockDim.x + threadIdx.x;
    if (t >= NPTS * 192) return;
    int n = t / 192, c = t - n * 192;
    _Float16 v;
    if (c < 64)       v = x1[(size_t)n * 64 + c];
    else if (c < 128) v = x2[(size_t)n * 64 + (c - 64)];
    else              v = x3[(size_t)n * 64 + (c - 128)];
    out[t] = v;
}

// ---------------------------------------------------------------------------
// Segment max over each point cloud with fused BN-affine + LeakyReLU.
// grid = NSEG*C blocks; block reduces SEGSZ rows via LDS.
// ---------------------------------------------------------------------------
__launch_bounds__(256)
__global__ void segmax_kernel(const _Float16* __restrict__ R,
                              const float* __restrict__ scale,
                              const float* __restrict__ shift,
                              float* __restrict__ pooled, int C) {
    int b = blockIdx.x / C;
    int c = blockIdx.x - b * C;
    float sc = scale[c], sh = shift[c];
    const _Float16* base = R + (size_t)b * SEGSZ * C + c;
    float mx = -3.4e38f;
    for (int r = threadIdx.x; r < SEGSZ; r += 256) {
        float v = lrelu(fmaf((float)base[(size_t)r * C], sc, sh));
        mx = fmaxf(mx, v);
    }
    __shared__ float red[256];
    red[threadIdx.x] = mx;
    __syncthreads();
    for (int s = 128; s > 0; s >>= 1) {
        if (threadIdx.x < s)
            red[threadIdx.x] = fmaxf(red[threadIdx.x], red[threadIdx.x + s]);
        __syncthreads();
    }
    if (threadIdx.x == 0) pooled[(size_t)b * C + c] = red[0];
}

// ---------------------------------------------------------------------------
// concat [pooled[batch[n]] | x1 | x2 | x3] -> [N,1216] f16
// ---------------------------------------------------------------------------
__global__ void concat7_kernel(const float* __restrict__ pooled,
                               const long long* __restrict__ batch,
                               const _Float16* __restrict__ x1,
                               const _Float16* __restrict__ x2,
                               const _Float16* __restrict__ x3,
                               _Float16* __restrict__ out) {
    int t = blockIdx.x * blockDim.x + threadIdx.x;
    if (t >= NPTS * 1216) return;
    int n = t / 1216, c = t - n * 1216;
    _Float16 v;
    if (c < 1024)      v = (_Float16)pooled[(size_t)batch[n] * 1024 + c];
    else if (c < 1088) v = x1[(size_t)n * 64 + (c - 1024)];
    else if (c < 1152) v = x2[(size_t)n * 64 + (c - 1088)];
    else               v = x3[(size_t)n * 64 + (c - 1152)];
    out[t] = v;
}

// ---------------------------------------------------------------------------
// Host-side orchestration
// ---------------------------------------------------------------------------
extern "C" void kernel_launch(void* const* d_in, const int* in_sizes, int n_in,
                              void* d_out, int out_size, void* d_ws, size_t ws_size,
                              hipStream_t stream) {
    (void)in_sizes; (void)n_in; (void)out_size; (void)ws_size;

    // -------- input pointers (setup_inputs dict order, params flattened) ----
    const float* feats = (const float*)d_in[0];
    // d_in[1] = xyz (unused by reference)
    const float* W1_0 = (const float*)d_in[2];
    const float* g1_0 = (const float*)d_in[3];  const float* b1_0 = (const float*)d_in[4];
    const float* W1_1 = (const float*)d_in[5];
    const float* g1_1 = (const float*)d_in[6];  const float* b1_1 = (const float*)d_in[7];
    const float* W2_0 = (const float*)d_in[8];
    const float* g2_0 = (const float*)d_in[9];  const float* b2_0 = (const float*)d_in[10];
    const float* W2_1 = (const float*)d_in[11];
    const float* g2_1 = (const float*)d_in[12]; const float* b2_1 = (const float*)d_in[13];
    const float* W3_0 = (const float*)d_in[14];
    const float* g3_0 = (const float*)d_in[15]; const float* b3_0 = (const float*)d_in[16];
    const float* W6   = (const float*)d_in[17];
    const float* g6   = (const float*)d_in[18]; const float* b6 = (const float*)d_in[19];
    const float* W7   = (const float*)d_in[20];
    const float* g7   = (const float*)d_in[21]; const float* b7 = (const float*)d_in[22];
    const float* W8   = (const float*)d_in[23];
    const float* g8   = (const float*)d_in[24]; const float* b8 = (const float*)d_in[25];
    const float* W9   = (const float*)d_in[26];
    // d_in[27] = offset (unused; segments are fixed-size)
    const long long* batch = (const long long*)d_in[28];
    const long long* nidx  = (const long long*)d_in[29];
    float* outFinal = (float*)d_out;

    // -------- workspace bump allocator (256B aligned) -----------------------
    char* ws = (char*)d_ws;
    size_t off = 0;
    auto alloc = [&](size_t bytes) -> char* {
        char* p = ws + off;
        off += (bytes + 255) & ~(size_t)255;
        return p;
    };

    _Float16* w1_0h = (_Float16*)alloc(64   * 32   * 2);
    _Float16* w1_1h = (_Float16*)alloc(64   * 64   * 2);
    _Float16* w2_0h = (_Float16*)alloc(64   * 128  * 2);
    _Float16* w2_1h = (_Float16*)alloc(64   * 64   * 2);
    _Float16* w3_0h = (_Float16*)alloc(64   * 128  * 2);
    _Float16* w6h   = (_Float16*)alloc(1024 * 192  * 2);
    _Float16* w7h   = (_Float16*)alloc(512  * 1216 * 2);
    _Float16* w8h   = (_Float16*)alloc(256  * 512  * 2);
    _Float16* w9h   = (_Float16*)alloc(16   * 256  * 2);

    // 8 BN layers, stride 1024 channels each; sum+sumsq contiguous for memset
    float* statsAcc  = (float*)alloc(2 * 8 * 1024 * 4);
    float* statSum   = statsAcc;
    float* statSq    = statsAcc + 8 * 1024;
    float* bnScale   = (float*)alloc(8 * 1024 * 4);
    float* bnShift   = (float*)alloc(8 * 1024 * 4);
    _Float16* bnScaleH = (_Float16*)alloc(8 * 1024 * 2);
    _Float16* bnShiftH = (_Float16*)alloc(8 * 1024 * 2);

    _Float16* x1h = (_Float16*)alloc((size_t)NPTS * 64 * 2);
    _Float16* x2h = (_Float16*)alloc((size_t)NPTS * 64 * 2);
    _Float16* x3h = (_Float16*)alloc((size_t)NPTS * 64 * 2);
    float* pooled = (float*)alloc(NSEG * 1024 * 4);

    // Big reusable regions
    _Float16* regA = (_Float16*)alloc((size_t)NPTS * 1216 * 2 > NKROWS * 128 * 2
                                      ? (size_t)NPTS * 1216 * 2 : NKROWS * 128 * 2);
    _Float16* regB = (_Float16*)alloc(NKROWS * 64 * 2);   // == N*1024*2
    _Float16* regC = (_Float16*)alloc(NKROWS * 64 * 2);

    // -------- helpers -------------------------------------------------------
    auto gemm = [&](const _Float16* A, const _Float16* Wh, _Float16* Ch, float* oF,
                    int layer, const _Float16* aScaleH, const _Float16* aShiftH,
                    int M, int Kc, int O, int Oreal) {
        long long blks = (long long)(M >> 5) * (O >> 4);
        int blocks = (int)((blks + 7) / 8);
        float* s  = (layer >= 0) ? statSum + layer * 1024 : nullptr;
        float* s2 = (layer >= 0) ? statSq  + layer * 1024 : nullptr;
        if (aScaleH)
            wmma_gemm_kernel<true><<<blocks, 256, 0, stream>>>(
                A, Wh, Ch, oF, s, s2, aScaleH, aShiftH, M, Kc, O, Oreal);
        else
            wmma_gemm_kernel<false><<<blocks, 256, 0, stream>>>(
                A, Wh, Ch, oF, s, s2, nullptr, nullptr, M, Kc, O, Oreal);
    };
    auto finalize = [&](int layer, const float* g, const float* b, int O, float invM) {
        bn_finalize_kernel<<<(O + 255) / 256, 256, 0, stream>>>(
            statSum + layer * 1024, statSq + layer * 1024, g, b,
            bnScale + layer * 1024, bnShift + layer * 1024,
            bnScaleH + layer * 1024, bnShiftH + layer * 1024, O, invM);
    };
    auto SC  = [&](int layer) { return bnScale  + layer * 1024; };
    auto SH  = [&](int layer) { return bnShift  + layer * 1024; };
    auto SCH = [&](int layer) { return bnScaleH + layer * 1024; };
    auto SHH = [&](int layer) { return bnShiftH + layer * 1024; };

    const float invNK = 1.0f / (float)NKROWS;
    const float invN  = 1.0f / (float)NPTS;

    // -------- weights -> f16 (padded) ---------------------------------------
    auto wc = [&](const float* W, _Float16* Wh, int Or, int Ir, int Op, int Ip) {
        int total = Op * Ip;
        wcvt_kernel<<<(total + 255) / 256, 256, 0, stream>>>(W, Wh, Or, Ir, Op, Ip);
    };
    wc(W1_0, w1_0h, 64, 12, 64, 32);
    wc(W1_1, w1_1h, 64, 64, 64, 64);
    wc(W2_0, w2_0h, 64, 128, 64, 128);
    wc(W2_1, w2_1h, 64, 64, 64, 64);
    wc(W3_0, w3_0h, 64, 128, 64, 128);
    wc(W6,   w6h, 1024, 192, 1024, 192);
    wc(W7,   w7h, 512, 1216, 512, 1216);
    wc(W8,   w8h, 256, 512, 256, 512);
    wc(W9,   w9h, 13, 256, 16, 256);

    hipMemsetAsync(statsAcc, 0, 2 * 8 * 1024 * sizeof(float), stream);

    // -------- edge conv 1 ----------------------------------------------------
    gather_first_kernel<<<(int)((NKROWS + 255) / 256), 256, 0, stream>>>(feats, nidx, regA);
    gemm(regA, w1_0h, regB, nullptr, 0, nullptr, nullptr, (int)NKROWS, 32, 64, 64);
    finalize(0, g1_0, b1_0, 64, invNK);
    gemm(regB, w1_1h, regC, nullptr, 1, SCH(0), SHH(0), (int)NKROWS, 64, 64, 64);
    finalize(1, g1_1, b1_1, 64, invNK);
    maxk_apply_kernel<<<(NPTS * 64 + 255) / 256, 256, 0, stream>>>(regC, SC(1), SH(1), x1h, 64);

    // -------- edge conv 2 ----------------------------------------------------
    gather_edge64_kernel<<<(int)((NKROWS * 8 + 255) / 256), 256, 0, stream>>>(x1h, nidx, regA);
    gemm(regA, w2_0h, regB, nullptr, 2, nullptr, nullptr, (int)NKROWS, 128, 64, 64);
    finalize(2, g2_0, b2_0, 64, invNK);
    gemm(regB, w2_1h, regC, nullptr, 3, SCH(2), SHH(2), (int)NKROWS, 64, 64, 64);
    finalize(3, g2_1, b2_1, 64, invNK);
    maxk_apply_kernel<<<(NPTS * 64 + 255) / 256, 256, 0, stream>>>(regC, SC(3), SH(3), x2h, 64);

    // -------- edge conv 3 ----------------------------------------------------
    gather_edge64_kernel<<<(int)((NKROWS * 8 + 255) / 256), 256, 0, stream>>>(x2h, nidx, regA);
    gemm(regA, w3_0h, regB, nullptr, 4, nullptr, nullptr, (int)NKROWS, 128, 64, 64);
    finalize(4, g3_0, b3_0, 64, invNK);
    maxk_apply_kernel<<<(NPTS * 64 + 255) / 256, 256, 0, stream>>>(regB, SC(4), SH(4), x3h, 64);

    // -------- conv6 [N,192] -> [N,1024] + segment max ------------------------
    concat3_kernel<<<(NPTS * 192 + 255) / 256, 256, 0, stream>>>(x1h, x2h, x3h, regA);
    gemm(regA, w6h, regB, nullptr, 5, nullptr, nullptr, NPTS, 192, 1024, 1024);
    finalize(5, g6, b6, 1024, invN);
    segmax_kernel<<<NSEG * 1024, 256, 0, stream>>>(regB, SC(5), SH(5), pooled, 1024);

    // -------- conv7/8/9 ------------------------------------------------------
    concat7_kernel<<<(NPTS * 1216 + 255) / 256, 256, 0, stream>>>(
        pooled, batch, x1h, x2h, x3h, regA);
    gemm(regA, w7h, regC, nullptr, 6, nullptr, nullptr, NPTS, 1216, 512, 512);
    finalize(6, g7, b7, 512, invN);
    gemm(regC, w8h, regB, nullptr, 7, SCH(6), SHH(6), NPTS, 512, 256, 256);
    finalize(7, g8, b8, 256, invN);
    gemm(regB, w9h, nullptr, outFinal, -1, SCH(7), SHH(7), NPTS, 256, 16, 13);
}
